// LSTM_Layer_74792560493067
// MI455X (gfx1250) — compile-verified, matrix operand order checked by reading, not run on hardware
//
#include <hip/hip_runtime.h>

// ---------------------------------------------------------------------------
// LSTM layer on MI455X (gfx1250), bf16 WMMA path, zero per-step conversion.
//   BATCH=64, SEQ=512, IN=512, UNITS=1024, D=IN+UNITS=1536
// One-time prep:  W (4 gates) f32 -> bf16 WMMA B-fragments (12.6 MB, L2)
//                 x            f32 -> bf16 row-major        (32 MB,  L2)
// Per timestep:   ONE fused kernel, 256 waves. Wave (mt,nt) owns a 16x16
//                 output tile, accumulates all 4 gates (A reused 4x) with a
//                 software-pipelined load->wmma loop split into x-region and
//                 h-region (all-immediate addressing, no per-iter selects),
//                 then applies the gate nonlinearities (HW tanh/exp),
//                 updates c (f32), writes h as bf16 (next step's A operand)
//                 and f32 to the output.
// ---------------------------------------------------------------------------

#define BATCH 64
#define SEQN  512
#define INDIM 512
#define UNITS 1024
#define DTOT  1536              // INDIM + UNITS
#define KSTEPS (DTOT/32)        // 48 WMMA k-steps (16 from x, 32 from h)
#define KX     16               // k-steps sourced from x
#define NTILES (UNITS/16)       // 64 n-tiles per gate
#define FRAG_ELEMS 512          // 32 lanes * 16 bf16 per fragment
#define WP_ELEMS  (4*NTILES*KSTEPS*FRAG_ELEMS)     // 6,291,456 bf16
#define WP_BYTES  ((size_t)WP_ELEMS*2)             // 12,582,912 B
#define X_ELEMS   ((size_t)BATCH*SEQN*INDIM)       // 16,777,216
#define X_BYTES   (X_ELEMS*2)                      // 33,554,432 B
#define H_ELEMS   (BATCH*UNITS)                    // 65,536

typedef __attribute__((ext_vector_type(16))) __bf16 v16bf;
typedef __attribute__((ext_vector_type(8)))  float  v8f;

union ABfrag { v16bf v; uint4 q[2]; };

__device__ __forceinline__ unsigned short f2bfu(float f) {
    union { float f; unsigned u; } in; in.f = f;
    unsigned b = in.u + 0x7FFFu + ((in.u >> 16) & 1u);   // round-nearest-even
    return (unsigned short)(b >> 16);
}

__device__ __forceinline__ float fast_exp(float v) {
#if __has_builtin(__builtin_amdgcn_exp2f)
    return __builtin_amdgcn_exp2f(v * 1.4426950408889634f);
#else
    return __expf(v);
#endif
}
__device__ __forceinline__ float fast_rcp(float v) {
#if __has_builtin(__builtin_amdgcn_rcpf)
    return __builtin_amdgcn_rcpf(v);
#else
    return 1.0f / v;
#endif
}
__device__ __forceinline__ float fast_sigmoid(float v) {
    return fast_rcp(1.0f + fast_exp(-v));
}
__device__ __forceinline__ float fast_tanh(float v) {
#if __has_builtin(__builtin_amdgcn_tanhf)
    return __builtin_amdgcn_tanhf(v);
#else
    float e = fast_exp(2.0f * v);
    return (e - 1.0f) * fast_rcp(e + 1.0f);
#endif
}

// Zero the bf16 h[0] buffer and the f32 c buffer (ws is poisoned / reused).
__global__ void init_state(unsigned short* __restrict__ hb0, float* __restrict__ cb) {
    int tid = blockIdx.x * blockDim.x + threadIdx.x;
    if (tid < H_ELEMS) hb0[tid] = 0;
    else if (tid < 2 * H_ELEMS) cb[tid - H_ELEMS] = 0.0f;
}

// x f32 -> bf16, row-major passthrough. 8 elements per thread.
__global__ void pack_x(const float4* __restrict__ x4, uint4* __restrict__ xb4) {
    size_t i = (size_t)blockIdx.x * blockDim.x + threadIdx.x;
    if (i >= X_ELEMS / 8) return;
    float4 a = x4[2 * i], b = x4[2 * i + 1];
    union { uint4 q; unsigned short s[8]; } o;
    o.s[0] = f2bfu(a.x); o.s[1] = f2bfu(a.y); o.s[2] = f2bfu(a.z); o.s[3] = f2bfu(a.w);
    o.s[4] = f2bfu(b.x); o.s[5] = f2bfu(b.y); o.s[6] = f2bfu(b.z); o.s[7] = f2bfu(b.w);
    xb4[i] = o.q;
}

// Pack W[g] (f32, [DTOT x UNITS] row-major) into bf16 WMMA B-fragments:
//   Wp[((g*NTILES+nt)*KSTEPS+kt)*32 + lane][16]
// element e maps to k_local = e + (e<8?0:8) + (lane<16?0:8),
// column n = nt*16 + (lane&15)   (mirrors the 16-bit A/B fragment layout).
__global__ void pack_weights(const float* __restrict__ Wf,
                             const float* __restrict__ Wi,
                             const float* __restrict__ Wc,
                             const float* __restrict__ Wo,
                             unsigned short* __restrict__ Wp) {
    int tid = blockIdx.x * blockDim.x + threadIdx.x;
    if (tid >= WP_ELEMS) return;
    int e    =  tid        & 15;
    int lane = (tid >> 4)  & 31;
    int kt   = (tid >> 9)  % KSTEPS;
    int rest =  tid / (512 * KSTEPS);
    int nt   = rest & (NTILES - 1);
    int g    = rest >> 6;
    const float* W = (g == 0) ? Wf : (g == 1) ? Wi : (g == 2) ? Wc : Wo;
    int klocal = e + ((e < 8) ? 0 : 8) + ((lane < 16) ? 0 : 8);
    int k = kt * 32 + klocal;
    int n = nt * 16 + (lane & 15);
    Wp[tid] = f2bfu(W[(size_t)k * UNITS + n]);
}

// One LSTM timestep, fully fused. 32 blocks x 256 threads = 256 waves.
__global__ __launch_bounds__(256) void lstm_step(
    const unsigned short* __restrict__ xb, const unsigned short* __restrict__ WpRaw,
    const float* __restrict__ bF, const float* __restrict__ bI,
    const float* __restrict__ bC, const float* __restrict__ bO,
    const unsigned short* __restrict__ hprev, unsigned short* __restrict__ hnext,
    float* __restrict__ cbuf, float* __restrict__ out, int t)
{
    const int lane = threadIdx.x & 31;
    const int wid  = (blockIdx.x * blockDim.x + threadIdx.x) >> 5; // 0..255
    const int mt   = wid & 3;        // batch tile (4 x 16 rows)
    const int nt   = wid >> 2;       // unit tile  (64 x 16 cols)

    // A-fragment addressing: lanes 0-15 -> rows 0-15 (K base +0),
    // lanes 16-31 -> same rows, K base +8.
    const int brow = mt * 16 + (lane & 15);
    const int koff = (lane < 16) ? 0 : 8;

    // Region bases (element units); kt-dependent offsets become immediates.
    const unsigned short* xbase = xb + ((size_t)brow * SEQN + t) * INDIM + koff;  // + kt*32, kt in [0,16)
    const unsigned short* hbase = hprev + (size_t)brow * UNITS - INDIM + koff;    // + kt*32, kt in [16,48)

    const v16bf* wp = (const v16bf*)WpRaw;
    const size_t fb0 = ((size_t)(0 * NTILES + nt) * KSTEPS) * 32 + lane;
    const size_t fb1 = ((size_t)(1 * NTILES + nt) * KSTEPS) * 32 + lane;
    const size_t fb2 = ((size_t)(2 * NTILES + nt) * KSTEPS) * 32 + lane;
    const size_t fb3 = ((size_t)(3 * NTILES + nt) * KSTEPS) * 32 + lane;

    auto load_frags = [&](const unsigned short* base, int kt,
                          ABfrag& a, v16bf& b0, v16bf& b1, v16bf& b2, v16bf& b3) {
        const unsigned short* src = base + kt * 32;
        a.q[0] = *(const uint4*)(src);        // k_local {0..7}+koff
        a.q[1] = *(const uint4*)(src + 16);   // k_local {16..23}+koff
        const size_t fo = (size_t)kt * 32;
        b0 = wp[fb0 + fo]; b1 = wp[fb1 + fo];
        b2 = wp[fb2 + fo]; b3 = wp[fb3 + fo];
    };

    v8f acc0 = {}, acc1 = {}, acc2 = {}, acc3 = {};

    auto do_wmma = [&](const ABfrag& a, const v16bf& b0, const v16bf& b1,
                       const v16bf& b2, const v16bf& b3) {
        acc0 = __builtin_amdgcn_wmma_f32_16x16x32_bf16(false, a.v, false, b0, (short)0, acc0, false, false);
        acc1 = __builtin_amdgcn_wmma_f32_16x16x32_bf16(false, a.v, false, b1, (short)0, acc1, false, false);
        acc2 = __builtin_amdgcn_wmma_f32_16x16x32_bf16(false, a.v, false, b2, (short)0, acc2, false, false);
        acc3 = __builtin_amdgcn_wmma_f32_16x16x32_bf16(false, a.v, false, b3, (short)0, acc3, false, false);
    };

    // Software pipeline (depth 1) across two flat-addressed regions.
    ABfrag a_c; v16bf b0_c, b1_c, b2_c, b3_c;
    ABfrag a_n; v16bf b0_n, b1_n, b2_n, b3_n;

    load_frags(xbase, 0, a_c, b0_c, b1_c, b2_c, b3_c);
#pragma unroll
    for (int kt = 0; kt < KX - 1; ++kt) {                 // kt = 0..14, x region
        load_frags(xbase, kt + 1, a_n, b0_n, b1_n, b2_n, b3_n);
        do_wmma(a_c, b0_c, b1_c, b2_c, b3_c);
        a_c = a_n; b0_c = b0_n; b1_c = b1_n; b2_c = b2_n; b3_c = b3_n;
    }
    // Bridge: prefetch first h-region fragment while consuming kt=15.
    load_frags(hbase, KX, a_n, b0_n, b1_n, b2_n, b3_n);
    do_wmma(a_c, b0_c, b1_c, b2_c, b3_c);
    a_c = a_n; b0_c = b0_n; b1_c = b1_n; b2_c = b2_n; b3_c = b3_n;
#pragma unroll
    for (int kt = KX; kt < KSTEPS - 1; ++kt) {            // kt = 16..46, h region
        load_frags(hbase, kt + 1, a_n, b0_n, b1_n, b2_n, b3_n);
        do_wmma(a_c, b0_c, b1_c, b2_c, b3_c);
        a_c = a_n; b0_c = b0_n; b1_c = b1_n; b2_c = b2_n; b3_c = b3_n;
    }
    do_wmma(a_c, b0_c, b1_c, b2_c, b3_c);                 // kt = 47

    // C/D layout: lane L holds column N = L%16, rows M = (L<16?0:8)+r.
    const int u     = nt * 16 + (lane & 15);
    const int rbase = (lane < 16) ? 0 : 8;
    const float bf_ = bF[u], bi_ = bI[u], bc_ = bC[u], bo_ = bO[u];
#pragma unroll
    for (int r = 0; r < 8; ++r) {
        const int bm = mt * 16 + rbase + r;
        float fg = fast_sigmoid(acc0[r] + bf_);
        float ig = fast_sigmoid(acc1[r] + bi_);
        float cc = fast_tanh   (acc2[r] + bc_);
        float og = fast_sigmoid(acc3[r] + bo_);
        const size_t ci = (size_t)bm * UNITS + u;
        float cn = fg * cbuf[ci] + ig * cc;
        float hn = og * fast_tanh(cn);
        cbuf[ci]  = cn;
        hnext[ci] = f2bfu(hn);
        out[((size_t)bm * SEQN + t) * UNITS + u] = hn;
    }
}

extern "C" void kernel_launch(void* const* d_in, const int* in_sizes, int n_in,
                              void* d_out, int out_size, void* d_ws, size_t ws_size,
                              hipStream_t stream) {
    (void)in_sizes; (void)n_in; (void)out_size; (void)ws_size;
    // setup_inputs order: x, W_f, W_i, W_o, W_c, b_f, b_i, b_o, b_c
    const float* x  = (const float*)d_in[0];
    const float* Wf = (const float*)d_in[1];
    const float* Wi = (const float*)d_in[2];
    const float* Wo = (const float*)d_in[3];
    const float* Wc = (const float*)d_in[4];
    const float* bF = (const float*)d_in[5];
    const float* bI = (const float*)d_in[6];
    const float* bO = (const float*)d_in[7];
    const float* bC = (const float*)d_in[8];
    float* out = (float*)d_out;

    char* ws = (char*)d_ws;
    unsigned short* Wp  = (unsigned short*)ws;                       // 12.6 MB
    unsigned short* xb  = (unsigned short*)(ws + WP_BYTES);          // 32 MB
    unsigned short* hb0 = (unsigned short*)(ws + WP_BYTES + X_BYTES);          // 128 KB
    unsigned short* hb1 = hb0 + H_ELEMS;                                       // 128 KB
    float*          cb  = (float*)(ws + WP_BYTES + X_BYTES + 4 * (size_t)H_ELEMS); // 256 KB

    init_state<<<(2 * H_ELEMS + 255) / 256, 256, 0, stream>>>(hb0, cb);
    pack_x<<<(int)((X_ELEMS / 8 + 255) / 256), 256, 0, stream>>>(
        (const float4*)x, (uint4*)xb);
    pack_weights<<<(WP_ELEMS + 255) / 256, 256, 0, stream>>>(Wf, Wi, Wc, Wo, Wp);

    for (int t = 0; t < SEQN; ++t) {
        const unsigned short* hp = (t & 1) ? hb1 : hb0;
        unsigned short*       hn = (t & 1) ? hb0 : hb1;
        lstm_step<<<32, 256, 0, stream>>>(xb, Wp, bF, bI, bC, bO, hp, hn, cb, out, t);
    }
}
